// DecoderWithAttention_64115271795113
// MI455X (gfx1250) — matrix operand order repaired and checked by reading, not compiled, not run on hardware
//
#include <hip/hip_runtime.h>
#include <hip/hip_bf16.h>
#include <cstdint>

// ---------------- problem constants ----------------
#define NUMC 97          // vocab
#define SOS  97          // SOS token id == NUM_CLASSES
#define EMBN 98          // NUM_CLASSES + 1
#define DD   512         // D (in_planes)
#define SS   512         // S (GRU hidden)
#define AA   512         // A (attDim)
#define TT   25          // steps / encoder length
#define BB   2048        // batch
#define G3S  1536        // 3*S
#define GK   1024        // D + A (GRU input width)
#define CPAD 128         // padded fc output dim

typedef __bf16 bf16;
typedef __attribute__((ext_vector_type(16))) __bf16 bf16x16;
typedef __attribute__((ext_vector_type(8)))  float  v8f;
typedef __attribute__((ext_vector_type(4)))  float  f32x4;

union FragAB { bf16x16 v; f32x4 f[2]; };
union FragC  { v8f v; float f[8]; };

// ---------------- pack / convert: f32 -> bf16, optional transpose, zero-pad rows ----------------
__global__ __launch_bounds__(256) void pack_bf16_kernel(
    const float* __restrict__ in, bf16* __restrict__ out,
    int N, int K, int ldin, int transpose, int Npad)
{
    long idx = (long)blockIdx.x * 256 + threadIdx.x;
    long total = (long)Npad * K;
    if (idx >= total) return;
    int n = (int)(idx / K), k = (int)(idx % K);
    float v = 0.f;
    if (n < N) v = transpose ? in[(size_t)k * ldin + n] : in[(size_t)n * ldin + k];
    out[idx] = (bf16)v;
}

__global__ __launch_bounds__(256) void init_s_kernel(float* sf, bf16* sb, long n)
{
    long idx = (long)blockIdx.x * 256 + threadIdx.x;
    if (idx >= n) return;
    sf[idx] = 0.f;
    sb[idx] = (bf16)0.f;
}

// ---------------- generic bf16 WMMA GEMM ----------------
// C[M,Nstore](f32 or bf16) = A[M,K](bf16, row-major, lda) x Bp[Npad,K](bf16, output-major) + bias
// Block tile 128x128 (8 wave32s: 4 in M x 2 in N), wave tile 32x64 (2x4 WMMA 16x16x32)
#define BM 128
#define BN 128
#define BK 32
#define LDSP 40   // padded row stride (halves) -> 80B, avoids LDS bank conflicts

__global__ __launch_bounds__(256) void gemm_bf16_kernel(
    const bf16* __restrict__ A, int lda,
    const bf16* __restrict__ Bp, int ldb,
    float* __restrict__ Cf, bf16* __restrict__ Cbf, int ldc,
    const float* __restrict__ bias,
    int M, int Nstore, int K)
{
    __shared__ __align__(16) bf16 As[2][BM * LDSP];
    __shared__ __align__(16) bf16 Bs[2][BN * LDSP];

    const int tid  = threadIdx.x;
    const int lane = tid & 31;
    const int wave = tid >> 5;
    const int m0 = blockIdx.x * BM;
    const int n0 = blockIdx.y * BN;
    const int wm = (wave >> 1) * 32;   // wave row offset in block tile
    const int wn = (wave & 1) * 64;    // wave col offset in block tile

    v8f c[2][4] = {};

    auto loadA = [&](int buf, int k0) {
        #pragma unroll
        for (int i = 0; i < 2; ++i) {
            int idx = tid + i * 256;          // 512 chunks of 8 halves = 128x32
            int row = idx >> 2;
            int kc  = (idx & 3) * 8;
            f32x4 v = *(const f32x4*)(A + (size_t)(m0 + row) * lda + k0 + kc);
            *(f32x4*)(&As[buf][row * LDSP + kc]) = v;
        }
    };
    auto loadB = [&](int buf, int k0) {
        #pragma unroll
        for (int i = 0; i < 2; ++i) {
            int idx = tid + i * 256;          // 512 chunks of 8 halves = 128x32
            int row = idx >> 2;
            int kc  = (idx & 3) * 8;
            f32x4 v = *(const f32x4*)(Bp + (size_t)(n0 + row) * ldb + k0 + kc);
            *(f32x4*)(&Bs[buf][row * LDSP + kc]) = v;
        }
    };

    loadA(0, 0);
    loadB(0, 0);
    __syncthreads();

    const int nk = K / BK;
    const int r = lane & 15, h = lane >> 4;

    for (int kk = 0; kk < nk; ++kk) {
        int buf = kk & 1;
        if (kk + 1 < nk) {
            loadA(buf ^ 1, (kk + 1) * BK);
            loadB(buf ^ 1, (kk + 1) * BK);
        }
        if (kk + 2 < nk)  // speculative L2 prefetch of the tile after next
            __builtin_prefetch(A + (size_t)(m0 + (tid >> 1)) * lda + (size_t)(kk + 2) * BK, 0, 1);

        FragAB a0, a1;
        {   // A fragments: lane holds row (wm+r), K halves [8h..8h+7] and [16+8h..23+8h]
            const bf16* p = &As[buf][(wm + r) * LDSP + 8 * h];
            a0.f[0] = *(const f32x4*)p;  a0.f[1] = *(const f32x4*)(p + 16);
            p = &As[buf][(wm + 16 + r) * LDSP + 8 * h];
            a1.f[0] = *(const f32x4*)p;  a1.f[1] = *(const f32x4*)(p + 16);
        }
        #pragma unroll
        for (int nt = 0; nt < 4; ++nt) {
            // B fragment: lane holds col (wn + 16*nt + r), K halves [8h..8h+7] and [16+8h..23+8h]
            FragAB b;
            const bf16* p = &Bs[buf][(wn + 16 * nt + r) * LDSP + 8 * h];
            b.f[0] = *(const f32x4*)p;  b.f[1] = *(const f32x4*)(p + 16);
            c[0][nt] = __builtin_amdgcn_wmma_f32_16x16x32_bf16(false, a0.v, false, b.v, (short)0, c[0][nt], false, false);
            c[1][nt] = __builtin_amdgcn_wmma_f32_16x16x32_bf16(false, a1.v, false, b.v, (short)0, c[1][nt], false, false);
        }

        __syncthreads();
    }

    // epilogue: C/D layout -> element (m,n): vgpr g = m%8, lane = (m/8)*16 + n
    const int cr = lane >> 4, cc = lane & 15;
    #pragma unroll
    for (int mt = 0; mt < 2; ++mt) {
        #pragma unroll
        for (int nt = 0; nt < 4; ++nt) {
            int col = n0 + wn + nt * 16 + cc;
            if (col >= Nstore) continue;
            float bv = bias ? bias[col] : 0.f;
            FragC u; u.v = c[mt][nt];
            #pragma unroll
            for (int g = 0; g < 8; ++g) {
                int row = m0 + wm + mt * 16 + g + 8 * cr;
                float val = u.f[g] + bv;
                if (Cbf) Cbf[(size_t)row * ldc + col] = (bf16)val;
                else     Cf [(size_t)row * ldc + col] = val;
            }
        }
    }
}

// ---------------- attention + gin assembly (one block per batch row) ----------------
__global__ __launch_bounds__(256) void attn_kernel(
    const bf16*  __restrict__ xProj,     // [B,T,A] bf16 (this decoder)
    const float* __restrict__ sProj,     // [B,A] f32
    const bf16*  __restrict__ xb16,      // [B,T,D] bf16 (shared; L2-resident)
    const float* __restrict__ wv,        // [A]
    const float* __restrict__ wvb,       // [1]
    const bf16*  __restrict__ embb,      // [98,A] bf16
    const int*   __restrict__ tgt,       // [B,T]
    bf16*        __restrict__ gin,       // [B, A+D] bf16
    int t)
{
    const int b    = blockIdx.x;
    const int tid  = threadIdx.x;
    const int lane = tid & 31;
    const int wave = tid >> 5;

    __shared__ float e_sh[32];

    // e[l] = tanh(sProj + xProj[l]) . wv + wvb   (one wave per l, looped)
    const float* sp = sProj + (size_t)b * AA;
    for (int l = wave; l < TT; l += 8) {
        const bf16* xp = xProj + ((size_t)b * TT + l) * AA;
        float partial = 0.f;
        for (int a = lane; a < AA; a += 32)
            partial += tanhf(sp[a] + (float)xp[a]) * wv[a];
        #pragma unroll
        for (int off = 16; off; off >>= 1) partial += __shfl_xor(partial, off, 32);
        if (lane == 0) e_sh[l] = partial + wvb[0];
    }
    __syncthreads();

    // softmax over L=25 in wave 0 (wave32 shuffle reduction)
    if (wave == 0) {
        float v = (lane < TT) ? e_sh[lane] : -3.4e38f;
        float m = v;
        #pragma unroll
        for (int off = 16; off; off >>= 1) m = fmaxf(m, __shfl_xor(m, off, 32));
        float ex = (lane < TT) ? __expf(v - m) : 0.f;
        float s = ex;
        #pragma unroll
        for (int off = 16; off; off >>= 1) s += __shfl_xor(s, off, 32);
        if (lane < TT) e_sh[lane] = ex / s;
    }
    __syncthreads();

    // ctx[d] = sum_l alpha[l] * x[b,l,d]  -> gin[b, A+d] (bf16)
    const bf16* xrow = xb16 + (size_t)b * TT * DD;
    bf16* gb = gin + (size_t)b * (AA + DD);
    for (int d = tid; d < DD; d += 256) {
        float acc = 0.f;
        #pragma unroll
        for (int l = 0; l < TT; ++l) acc += e_sh[l] * (float)xrow[(size_t)l * DD + d];
        gb[AA + d] = (bf16)acc;
    }

    // embedding gather -> gin[b, 0:A]
    int yprev = (t == 0) ? SOS : tgt[(size_t)b * TT + (t - 1)];
    const bf16* ep = embb + (size_t)yprev * AA;
    for (int a = tid; a < AA; a += 256) gb[a] = ep[a];
}

// ---------------- fused GRU gates + state update ----------------
__global__ __launch_bounds__(256) void gates_kernel(
    const float* __restrict__ gi,  // [B,3S]
    const float* __restrict__ gh,  // [B,3S]
    float* __restrict__ sf,        // [B,S] f32 master state
    bf16*  __restrict__ sb)        // [B,S] bf16 state for GEMMs
{
    long idx = (long)blockIdx.x * 256 + threadIdx.x;     // B*S threads
    int b = (int)(idx >> 9), j = (int)(idx & 511);
    size_t base = (size_t)b * G3S;
    float ir = gi[base + j], iz = gi[base + SS + j], inn = gi[base + 2 * SS + j];
    float hr = gh[base + j], hz = gh[base + SS + j], hn  = gh[base + 2 * SS + j];
    float rg = 1.f / (1.f + __expf(-(ir + hr)));
    float zg = 1.f / (1.f + __expf(-(iz + hz)));
    float ng = tanhf(inn + rg * hn);
    float s  = sf[idx];
    float sn = (1.f - zg) * ng + zg * s;
    sf[idx] = sn;
    sb[idx] = (bf16)sn;
}

// ---------------- host orchestration ----------------
static inline void* bump(char*& p, size_t bytes) {
    void* r = (void*)p;
    p += (bytes + 255) & ~(size_t)255;
    return r;
}

extern "C" void kernel_launch(void* const* d_in, const int* in_sizes, int n_in,
                              void* d_out, int out_size, void* d_ws, size_t ws_size,
                              hipStream_t stream) {
    (void)in_sizes; (void)n_in; (void)out_size; (void)ws_size;

    const float* x       = (const float*)d_in[0];
    const int*   targets = (const int*)  d_in[1];
    const float* Wx  = (const float*)d_in[2];
    const float* bx  = (const float*)d_in[3];
    const float* Ws  = (const float*)d_in[4];
    const float* bs  = (const float*)d_in[5];
    const float* wv  = (const float*)d_in[6];
    const float* wvb = (const float*)d_in[7];
    const float* emb = (const float*)d_in[8];
    const float* Wih = (const float*)d_in[9];
    const float* bih = (const float*)d_in[10];
    const float* Whh = (const float*)d_in[11];
    const float* bhh = (const float*)d_in[12];
    const float* Wfc = (const float*)d_in[13];
    const float* bfc = (const float*)d_in[14];
    float* out = (float*)d_out;

    char* ws = (char*)d_ws;
    bf16* x_bf     = (bf16*)bump(ws, (size_t)BB * TT * DD * 2);
    bf16* xproj_bf = (bf16*)bump(ws, (size_t)2 * BB * TT * AA * 2);
    bf16* WxT_bf   = (bf16*)bump(ws, (size_t)2 * AA * DD * 2);
    bf16* WsT_bf   = (bf16*)bump(ws, (size_t)2 * AA * SS * 2);
    bf16* Wih_bf   = (bf16*)bump(ws, (size_t)2 * G3S * GK * 2);
    bf16* Whh_bf   = (bf16*)bump(ws, (size_t)2 * G3S * SS * 2);
    bf16* Wfc_bf   = (bf16*)bump(ws, (size_t)2 * CPAD * SS * 2);
    bf16* emb_bf   = (bf16*)bump(ws, (size_t)2 * EMBN * AA * 2);
    float* s_f32   = (float*)bump(ws, (size_t)2 * BB * SS * 4);
    bf16*  s_bf    = (bf16*) bump(ws, (size_t)2 * BB * SS * 2);
    float* sproj   = (float*)bump(ws, (size_t)2 * BB * AA * 4);
    bf16*  gin     = (bf16*) bump(ws, (size_t)2 * BB * GK * 2);
    float* gi_ws   = (float*)bump(ws, (size_t)2 * BB * G3S * 4);
    float* gh_ws   = (float*)bump(ws, (size_t)2 * BB * G3S * 4);

    auto packGrid = [](long n) { return dim3((unsigned)((n + 255) / 256)); };

    // ---- one-time packs (deterministic; re-run every call) ----
    for (int d = 0; d < 2; ++d) {
        pack_bf16_kernel<<<packGrid((long)AA * DD), 256, 0, stream>>>(
            Wx + (size_t)d * DD * AA, WxT_bf + (size_t)d * AA * DD, AA, DD, AA, 1, AA);
        pack_bf16_kernel<<<packGrid((long)AA * SS), 256, 0, stream>>>(
            Ws + (size_t)d * SS * AA, WsT_bf + (size_t)d * AA * SS, AA, SS, AA, 1, AA);
        pack_bf16_kernel<<<packGrid((long)G3S * GK), 256, 0, stream>>>(
            Wih + (size_t)d * G3S * GK, Wih_bf + (size_t)d * G3S * GK, G3S, GK, GK, 0, G3S);
        pack_bf16_kernel<<<packGrid((long)G3S * SS), 256, 0, stream>>>(
            Whh + (size_t)d * G3S * SS, Whh_bf + (size_t)d * G3S * SS, G3S, SS, SS, 0, G3S);
        pack_bf16_kernel<<<packGrid((long)CPAD * SS), 256, 0, stream>>>(
            Wfc + (size_t)d * NUMC * SS, Wfc_bf + (size_t)d * CPAD * SS, NUMC, SS, SS, 0, CPAD);
        pack_bf16_kernel<<<packGrid((long)EMBN * AA), 256, 0, stream>>>(
            emb + (size_t)d * EMBN * AA, emb_bf + (size_t)d * EMBN * AA, EMBN, AA, AA, 0, EMBN);
    }
    pack_bf16_kernel<<<packGrid((long)BB * TT * DD), 256, 0, stream>>>(
        x, x_bf, BB * TT, DD, DD, 0, BB * TT);
    init_s_kernel<<<packGrid((long)2 * BB * SS), 256, 0, stream>>>(s_f32, s_bf, (long)2 * BB * SS);

    // ---- xProj = x @ Wx + bx (bf16 output, reused by all 25 steps) ----
    for (int d = 0; d < 2; ++d) {
        gemm_bf16_kernel<<<dim3(BB * TT / BM, AA / BN), 256, 0, stream>>>(
            x_bf, DD, WxT_bf + (size_t)d * AA * DD, DD,
            nullptr, xproj_bf + (size_t)d * BB * TT * AA, AA,
            bx + (size_t)d * AA, BB * TT, AA, DD);
    }

    // ---- recurrent decode, teacher forcing ----
    for (int t = 0; t < TT; ++t) {
        for (int d = 0; d < 2; ++d) {
            bf16*  sb = s_bf  + (size_t)d * BB * SS;
            float* sf = s_f32 + (size_t)d * BB * SS;
            float* sp = sproj + (size_t)d * BB * AA;
            bf16*  gb = gin   + (size_t)d * BB * GK;
            float* gi = gi_ws + (size_t)d * BB * G3S;
            float* gh = gh_ws + (size_t)d * BB * G3S;

            // sProj = s @ Ws + bs
            gemm_bf16_kernel<<<dim3(BB / BM, AA / BN), 256, 0, stream>>>(
                sb, SS, WsT_bf + (size_t)d * AA * SS, SS,
                sp, nullptr, AA, bs + (size_t)d * AA, BB, AA, SS);

            // attention + gin = [emb[y_prev], ctx]
            attn_kernel<<<dim3(BB), 256, 0, stream>>>(
                xproj_bf + (size_t)d * BB * TT * AA, sp, x_bf,
                wv + (size_t)d * AA, wvb + d,
                emb_bf + (size_t)d * EMBN * AA,
                targets + (size_t)d * BB * TT, gb, t);

            // gi = gin @ Wih^T + bih
            gemm_bf16_kernel<<<dim3(BB / BM, G3S / BN), 256, 0, stream>>>(
                gb, GK, Wih_bf + (size_t)d * G3S * GK, GK,
                gi, nullptr, G3S, bih + (size_t)d * G3S, BB, G3S, GK);

            // gh = s @ Whh^T + bhh  (uses OLD state)
            gemm_bf16_kernel<<<dim3(BB / BM, G3S / BN), 256, 0, stream>>>(
                sb, SS, Whh_bf + (size_t)d * G3S * SS, SS,
                gh, nullptr, G3S, bhh + (size_t)d * G3S, BB, G3S, SS);

            // gates -> new s (f32 + bf16)
            gates_kernel<<<dim3(BB * SS / 256), 256, 0, stream>>>(gi, gh, sf, sb);

            // logits = s_new @ Wfc^T + bfc  -> d_out[d, :, t, :]
            gemm_bf16_kernel<<<dim3(BB / BM, CPAD / BN), 256, 0, stream>>>(
                sb, SS, Wfc_bf + (size_t)d * CPAD * SS, SS,
                out + (size_t)d * BB * TT * NUMC + (size_t)t * NUMC, nullptr, TT * NUMC,
                bfc + (size_t)d * NUMC, BB, NUMC, SS);
        }
    }
}
